// FMSlate_39659728011358
// MI455X (gfx1250) — matrix operand order repaired and checked by reading, not compiled, not run on hardware
//
#include <hip/hip_runtime.h>
#include <math.h>

typedef __attribute__((ext_vector_type(16))) _Float16 v16h;
typedef __attribute__((ext_vector_type(8)))  float    v8f;
typedef int v4i __attribute__((vector_size(16)));

#define S_SLATE 50
#define DEMB 16
#define WAVES_PER_BLOCK 4
#define ELEMS_PER_WAVE 16

#if defined(__AMDGCN__) && __has_builtin(__builtin_amdgcn_global_load_async_to_lds_b128)
#define HAS_ASYNC_COPY 1
#else
#define HAS_ASYNC_COPY 0
#endif

// Per-wave LDS scratch (~15.6 KB); 4 waves -> ~63.7 KB static LDS.
struct alignas(16) WaveShm {
  float    a32[64 * 16];     // slate rows (f32, async-copied), padded to 64 rows
  float    kbuf[64 * 16];    // k projection (f32: hot score path, no cvt)
  _Float16 vbuf[64 * 16];    // v projection (f16)
  float    sbuf[4 * 64];     // per-head scores / softmax weights
  float    q16[16];          // q row
  _Float16 tA[16 * 32];      // [uE | att] then [uE | teacher] input tile
  _Float16 hid[16 * 16];     // hidden activations between layers
  float    uEf[16 * 16];     // f32 uE for FM
  float    iEf[16 * 16];     // f32 iE for FM
  float    outf[16 * 16];    // decoder output (input_slate), f32
};

// Wave-level ordering fence: intra-wave LDS ops complete in order (DScnt is
// in-order for LDS per ISA §7.3), so each wave only needs a scheduling fence,
// not a workgroup barrier, for its private LDS region.
__device__ __forceinline__ void wavefence() { __builtin_amdgcn_wave_barrier(); }

#if HAS_ASYNC_COPY
__device__ __forceinline__ void async_copy16(const void* g, void* l) {
  // GLOBAL_LOAD_ASYNC_TO_LDS_B128: 16B per lane, memory -> LDS, ASYNCcnt.
  // Builtin prototype: (v4i AS1*, v4i AS3*, imm offset, imm cpol).
  __builtin_amdgcn_global_load_async_to_lds_b128(
      (__attribute__((address_space(1))) v4i*)g,
      (__attribute__((address_space(3))) v4i*)l, 0, 0);
}
__device__ __forceinline__ void wait_async0() {
#if __has_builtin(__builtin_amdgcn_s_wait_asynccnt)
  __builtin_amdgcn_s_wait_asynccnt(0);
#else
  asm volatile("s_wait_asynccnt 0x0" ::: "memory");
#endif
}
#endif

__device__ __forceinline__ v8f wmma16(v16h a, v16h b, v8f c) {
  return __builtin_amdgcn_wmma_f32_16x16x32_f16(
      /*neg_a=*/false, a, /*neg_b=*/false, b,
      /*c_mod=*/(short)0, c, /*reuse_a=*/false, /*reuse_b=*/false);
}

// A fragment from an f32 LDS tile, actual K=16, zero-padded to 32.
// Per ISA 16-bit A 16x32 layout: lane kh=lane>>4 holds K = kh*8+j (elems 0..7)
// and K = 16+kh*8+j (elems 8..15).
__device__ __forceinline__ v16h ldA_k16_f32(const float* t, int stride) {
  const int lane = threadIdx.x & 31;
  const int row = lane & 15, kh = (lane >> 4) & 1;
  v16h a;
#pragma unroll
  for (int j = 0; j < 8; ++j) {
    a[j]     = (_Float16)t[row * stride + kh * 8 + j];
    a[8 + j] = (_Float16)0.0f;
  }
  return a;
}

// A fragment from an f16 LDS tile, actual K=16 zero-padded.
__device__ __forceinline__ v16h ldA_k16(const _Float16* t, int stride) {
  const int lane = threadIdx.x & 31;
  const int row = lane & 15, kh = (lane >> 4) & 1;
  v16h a;
#pragma unroll
  for (int j = 0; j < 8; ++j) {
    a[j]     = t[row * stride + kh * 8 + j];
    a[8 + j] = (_Float16)0.0f;
  }
  return a;
}

__device__ __forceinline__ v16h ldA_k32(const _Float16* t, int stride) {
  const int lane = threadIdx.x & 31;
  const int row = lane & 15, kh = (lane >> 4) & 1;
  v16h a;
#pragma unroll
  for (int j = 0; j < 8; ++j) {
    a[j]     = t[row * stride + kh * 8 + j];
    a[8 + j] = t[row * stride + 16 + kh * 8 + j];
  }
  return a;
}

// B fragment from row-major f32 weight W[K x 16]; lane kh=lane>>4 holds K = kh*8+j
// (elems 0..7) and K = 16+kh*8+j (elems 8..15).  K beyond Kdim zero-padded.
__device__ __forceinline__ v16h ldB_w(const float* W, int Kdim) {
  const int lane = threadIdx.x & 31;
  const int col = lane & 15, kh = (lane >> 4) & 1;
  v16h b;
#pragma unroll
  for (int j = 0; j < 16; ++j) {
    const int k = (j < 8) ? kh * 8 + j : 16 + kh * 8 + (j - 8);
    b[j] = (k < Kdim) ? (_Float16)W[k * 16 + col] : (_Float16)0.0f;
  }
  return b;
}

__global__ void init_aux_kernel(float* aux) {
  if (threadIdx.x == 0 && blockIdx.x == 0) *aux = 0.0f;
}

__global__ __launch_bounds__(WAVES_PER_BLOCK * 32, 1) void fmslate_fused(
    const int* __restrict__ users, const int* __restrict__ items,
    const int* __restrict__ slate_ids,
    const float* __restrict__ user_emb, const float* __restrict__ item_emb,
    const float* __restrict__ user_lin, const float* __restrict__ item_lin,
    const float* __restrict__ slate_emb,
    const float* __restrict__ es_w1, const float* __restrict__ es_b1,
    const float* __restrict__ es_w2, const float* __restrict__ es_b2,
    const float* __restrict__ et_w1, const float* __restrict__ et_b1,
    const float* __restrict__ et_w2, const float* __restrict__ et_b2,
    const float* __restrict__ dec_w1, const float* __restrict__ dec_b1,
    const float* __restrict__ dec_w2, const float* __restrict__ dec_b2,
    const float* __restrict__ q_w, const float* __restrict__ q_b,
    const float* __restrict__ k_w, const float* __restrict__ k_b,
    const float* __restrict__ v_w, const float* __restrict__ v_b,
    const float* __restrict__ st_w, const float* __restrict__ st_b,
    float* __restrict__ out, float* __restrict__ aux, float invB) {
  __shared__ WaveShm shm[WAVES_PER_BLOCK];

  const int lane = threadIdx.x & 31;
  const int wid  = threadIdx.x >> 5;
  const int n    = lane & 15;          // D-frag / B-frag column
  const int kh   = (lane >> 4) & 1;    // half-select
  const int idxBase = blockIdx.x * (WAVES_PER_BLOCK * ELEMS_PER_WAVE) + wid * ELEMS_PER_WAVE;
  WaveShm& W = shm[wid];

  // ---- per-lane constants: column biases, q weights, weight B-fragments ----
  const float qb = q_b[n], kb = k_b[n], vb = v_b[n];
  const float esb1 = es_b1[n], esb2 = es_b2[n];
  const float etb1 = et_b1[n], etb2 = et_b2[n];
  const float db1 = dec_b1[n], db2 = dec_b2[n];
  float qwreg[16];
#pragma unroll
  for (int d = 0; d < 16; ++d) qwreg[d] = q_w[d * 16 + n];

  const v16h BkW  = ldB_w(k_w, 16),   BvW  = ldB_w(v_w, 16);
  const v16h Bes1 = ldB_w(es_w1, 16), Bes2 = ldB_w(es_w2, 16);
  const v16h Bet1 = ldB_w(et_w1, 32), Bet2 = ldB_w(et_w2, 16);
  const v16h Bd1  = ldB_w(dec_w1, 32), Bd2 = ldB_w(dec_w2, 16);
  const v8f zero = {};

  // ================= Phase 1: attention, one element at a time =============
  for (int e = 0; e < ELEMS_PER_WAVE; ++e) {
    const int idx = idxBase + e;
    const int item = items[idx];

    // q = targetS @ q_w + q_b (1x16 row, scalar via cross-lane shuffle)
    float ts = (lane < 16) ? slate_emb[(size_t)item * DEMB + lane] : 0.0f;
    float qn = qb;
#pragma unroll
    for (int d = 0; d < 16; ++d) qn = fmaf(__shfl(ts, d, 32), qwreg[d], qn);
    if (lane < 16) W.q16[lane] = qn;

    // ---- gather 50 slate rows (f32) into LDS ----
#if HAS_ASYNC_COPY
    {
      // 4 lanes per row, 16B per lane, straight to LDS (no VGPR staging).
      const int row8 = lane >> 2, qsel = lane & 3;
#pragma unroll
      for (int g = 0; g < 7; ++g) {
        const int t = g * 8 + row8;
        if (t < S_SLATE) {
          const int sid = slate_ids[idx * S_SLATE + t];
          async_copy16(slate_emb + (size_t)sid * DEMB + qsel * 4,
                       &W.a32[t * 16 + qsel * 4]);
        }
      }
    }
#else
    for (int t = lane; t < S_SLATE; t += 32) {
      const int sid = slate_ids[idx * S_SLATE + t];
      const float4* p = reinterpret_cast<const float4*>(slate_emb + (size_t)sid * DEMB);
      float4* d = reinterpret_cast<float4*>(&W.a32[t * 16]);
      d[0] = p[0]; d[1] = p[1]; d[2] = p[2]; d[3] = p[3];
    }
#endif
    // zero-pad rows 50..63
    for (int t = S_SLATE + lane; t < 64; t += 32) {
      const float4 z4 = {0.0f, 0.0f, 0.0f, 0.0f};
      float4* d = reinterpret_cast<float4*>(&W.a32[t * 16]);
      d[0] = z4; d[1] = z4; d[2] = z4; d[3] = z4;
    }
    // prefetch next element's first 16 slate rows
    if (e + 1 < ELEMS_PER_WAVE && lane < 16) {
      const int nsid = slate_ids[(idx + 1) * S_SLATE + lane];
      __builtin_prefetch(slate_emb + (size_t)nsid * DEMB, 0, 0);
    }
#if HAS_ASYNC_COPY
    wait_async0();
#endif
    wavefence();

    // k,v projections: 4 tiles x 2 WMMAs, bias added on D-fragment
#pragma unroll
    for (int tt = 0; tt < 4; ++tt) {
      const v16h a = ldA_k16_f32(&W.a32[tt * 16 * 16], 16);
      v8f kk = wmma16(a, BkW, zero);
      v8f vv = wmma16(a, BvW, zero);
#pragma unroll
      for (int r = 0; r < 8; ++r) {
        const int m = tt * 16 + r + 8 * kh;
        W.kbuf[m * 16 + n] = kk[r] + kb;
        W.vbuf[m * 16 + n] = (_Float16)(vv[r] + vb);
      }
    }
    wavefence();

    // scores s[h][t] = (q_h . k_h[t]) / 2   (k row reads are 16B-contiguous)
    for (int t = lane; t < S_SLATE; t += 32) {
      const float* kr = &W.kbuf[t * 16];
#pragma unroll
      for (int h = 0; h < 4; ++h) {
        float s = 0.0f;
#pragma unroll
        for (int j = 0; j < 4; ++j) s = fmaf(W.q16[4 * h + j], kr[4 * h + j], s);
        W.sbuf[h * 64 + t] = 0.5f * s;
      }
    }
    wavefence();

    // softmax over t per head (4 lanes)
    if (lane < 4) {
      float m = -1e30f;
      for (int t = 0; t < S_SLATE; ++t) m = fmaxf(m, W.sbuf[lane * 64 + t]);
      float s = 0.0f;
      for (int t = 0; t < S_SLATE; ++t) {
        const float ev = __expf(W.sbuf[lane * 64 + t] - m);
        W.sbuf[lane * 64 + t] = ev; s += ev;
      }
      const float inv = 1.0f / s;
      for (int t = 0; t < S_SLATE; ++t) W.sbuf[lane * 64 + t] *= inv;
    }
    wavefence();

    // attention output -> columns 16..31 of teacher input tile (f16)
    if (lane < 16) {
      const int h = lane >> 2;
      float acc = 0.0f;
      for (int t = 0; t < S_SLATE; ++t)
        acc = fmaf(W.sbuf[h * 64 + t], (float)W.vbuf[t * 16 + lane], acc);
      W.tA[e * 32 + 16 + lane] = (_Float16)acc;
    }
    wavefence();
  }

  // ================= Phase 2: MLP stack on 16-row tiles =====================
  {
    const int e2 = lane >> 1, half = lane & 1;
    const int idx2 = idxBase + e2;
    const int u = users[idx2], it2 = items[idx2];
    const float* up = user_emb + (size_t)u * DEMB + half * 8;
    const float* ip = item_emb + (size_t)it2 * DEMB + half * 8;
#pragma unroll
    for (int j = 0; j < 8; ++j) {
      const float uv = up[j], iv = ip[j];
      W.uEf[e2 * 16 + half * 8 + j] = uv;
      W.iEf[e2 * 16 + half * 8 + j] = iv;
      W.tA[e2 * 32 + half * 8 + j] = (_Float16)uv;  // cols 0..15 = uE
    }
  }
  wavefence();

  // student: relu(uE @ es_w1 + b1) @ es_w2 + b2
  v16h a = ldA_k16(&W.tA[0], 32);
  v8f h1 = wmma16(a, Bes1, zero);
#pragma unroll
  for (int r = 0; r < 8; ++r)
    W.hid[(r + 8 * kh) * 16 + n] = (_Float16)fmaxf(h1[r] + esb1, 0.0f);
  wavefence();
  a = ldA_k16(&W.hid[0], 16);
  v8f stf = wmma16(a, Bes2, zero);
#pragma unroll
  for (int r = 0; r < 8; ++r) stf[r] += esb2;

  // teacher: relu([uE|att] @ et_w1 + b1) @ et_w2 + b2   (K=32 exact)
  a = ldA_k32(&W.tA[0], 32);
  h1 = wmma16(a, Bet1, zero);
  wavefence();
#pragma unroll
  for (int r = 0; r < 8; ++r)
    W.hid[(r + 8 * kh) * 16 + n] = (_Float16)fmaxf(h1[r] + etb1, 0.0f);
  wavefence();
  a = ldA_k16(&W.hid[0], 16);
  v8f tef = wmma16(a, Bet2, zero);
#pragma unroll
  for (int r = 0; r < 8; ++r) tef[r] += etb2;

  // decoder: [uE|teacher] @ dec_w1 -> relu -> dec_w2   (K=32 exact)
#pragma unroll
  for (int r = 0; r < 8; ++r) W.tA[(r + 8 * kh) * 32 + 16 + n] = (_Float16)tef[r];
  wavefence();
  a = ldA_k32(&W.tA[0], 32);
  h1 = wmma16(a, Bd1, zero);
  wavefence();
#pragma unroll
  for (int r = 0; r < 8; ++r)
    W.hid[(r + 8 * kh) * 16 + n] = (_Float16)fmaxf(h1[r] + db1, 0.0f);
  wavefence();
  a = ldA_k16(&W.hid[0], 16);
  v8f sl = wmma16(a, Bd2, zero);
#pragma unroll
  for (int r = 0; r < 8; ++r) W.outf[(r + 8 * kh) * 16 + n] = sl[r] + db2;
  wavefence();

  // ================= Final: FM + sigmoid + aux reductions ===================
  float vpart = 0.0f;
#pragma unroll
  for (int r = 0; r < 8; ++r) {           // reg_loss partial (all lanes)
    const float d = stf[r] - tef[r];
    vpart = fmaf(d, d, vpart);
  }
  if (lane < 16) {
    const int e = lane, idx = idxBase + e;
    const float uLv = user_lin[users[idx]];
    const float iLv = item_lin[items[idx]];
    float fm = 0.0f;                      // 2-field FM term == dot(uE, iE)
#pragma unroll
    for (int d = 0; d < 16; ++d) fm = fmaf(W.uEf[e * 16 + d], W.iEf[e * 16 + d], fm);
    float slate = st_b[0];
#pragma unroll
    for (int j = 0; j < 16; ++j) slate = fmaf(W.outf[e * 16 + j], st_w[j], slate);
    const float z = uLv + iLv + fm + slate;
    out[idx] = 1.0f / (1.0f + __expf(-z));
    const float qq = fm + slate;
    vpart = fmaf(0.1f * qq, qq, vpart);
  }
#pragma unroll
  for (int off = 16; off > 0; off >>= 1) vpart += __shfl_down(vpart, off, 32);
  if (lane == 0) atomicAdd(aux, vpart * invB);
}

extern "C" void kernel_launch(void* const* d_in, const int* in_sizes, int n_in,
                              void* d_out, int out_size, void* d_ws, size_t ws_size,
                              hipStream_t stream) {
  const int*   users     = (const int*)d_in[0];
  const int*   items     = (const int*)d_in[1];
  const int*   slate_ids = (const int*)d_in[2];
  // d_in[3] slate_poses, d_in[4] slate_ratings: unused by reference
  const float* user_emb  = (const float*)d_in[5];
  const float* item_emb  = (const float*)d_in[6];
  const float* user_lin  = (const float*)d_in[7];
  const float* item_lin  = (const float*)d_in[8];
  const float* slate_emb = (const float*)d_in[9];
  const float* es_w1 = (const float*)d_in[10]; const float* es_b1 = (const float*)d_in[11];
  const float* es_w2 = (const float*)d_in[12]; const float* es_b2 = (const float*)d_in[13];
  const float* et_w1 = (const float*)d_in[14]; const float* et_b1 = (const float*)d_in[15];
  const float* et_w2 = (const float*)d_in[16]; const float* et_b2 = (const float*)d_in[17];
  const float* dec_w1 = (const float*)d_in[18]; const float* dec_b1 = (const float*)d_in[19];
  const float* dec_w2 = (const float*)d_in[20]; const float* dec_b2 = (const float*)d_in[21];
  const float* q_w = (const float*)d_in[22]; const float* q_b = (const float*)d_in[23];
  const float* k_w = (const float*)d_in[24]; const float* k_b = (const float*)d_in[25];
  const float* v_w = (const float*)d_in[26]; const float* v_b = (const float*)d_in[27];
  const float* st_w = (const float*)d_in[28]; const float* st_b = (const float*)d_in[29];

  const int B = in_sizes[0];               // users is (B,1)
  float* out = (float*)d_out;
  float* aux = out + B;                    // tuple output: (B,1) output then scalar aux

  init_aux_kernel<<<1, 32, 0, stream>>>(aux);

  const int elemsPerBlock = WAVES_PER_BLOCK * ELEMS_PER_WAVE;  // 64
  const int grid = (B + elemsPerBlock - 1) / elemsPerBlock;
  fmslate_fused<<<grid, WAVES_PER_BLOCK * 32, 0, stream>>>(
      users, items, slate_ids,
      user_emb, item_emb, user_lin, item_lin, slate_emb,
      es_w1, es_b1, es_w2, es_b2,
      et_w1, et_b1, et_w2, et_b2,
      dec_w1, dec_b1, dec_w2, dec_b2,
      q_w, q_b, k_w, k_b, v_w, v_b, st_w, st_b,
      out, aux, 1.0f / (float)B);
}
